// MultiHeadAttentionQuantum_65481071399990
// MI455X (gfx1250) — matrix operand order repaired and checked by reading, not compiled, not run on hardware
//
#include <hip/hip_runtime.h>
#include <hip/hip_bf16.h>

// ---------------------------------------------------------------------------
// Fused  y[b,i,n] = sum_j cos(x[b, s(i,j), f(i,j)] + theta[j&7]) * W[n,j]
// where the head-transpose in the reference is a pure layout reinterpretation:
// for an M-tile of 128 aligned rows, the 128x512 A-tile is the contiguous
// reinterpretation of x[b, :, mTile*8 : mTile*8+8].
//
// GEMM: M=65536 (B*S), N=512, K=512, fp32 in/out.
// Strategy: f16 WMMA (16x16x32) with fp32 accumulate, split-precision
// (Ahi*Whi + Ahi*Wlo + Alo*Whi) for ~fp32 accuracy at matrix-core speed.
// ---------------------------------------------------------------------------

typedef _Float16 v16h __attribute__((ext_vector_type(16)));
typedef _Float16 v8h  __attribute__((ext_vector_type(8)));
typedef float    v8f  __attribute__((ext_vector_type(8)));
typedef float    v4f  __attribute__((ext_vector_type(4)));

constexpr int kB  = 8;
constexpr int kS  = 8192;
constexpr int kE  = 512;    // 64 heads * 8
constexpr int kDk = 8;

constexpr int BM = 128;
constexpr int BN = 64;
constexpr int BK = 32;
constexpr int KTILES = kE / BK;   // 16
constexpr int LDA = 40;           // f16 elems/row -> 80B pitch (16B aligned, bank-friendly)
constexpr int LDB = 40;

__device__ __forceinline__ v8f wmma_f16(v16h a, v16h b, v8f c) {
    return __builtin_amdgcn_wmma_f32_16x16x32_f16(
        /*neg_a=*/false, a, /*neg_b=*/false, b,
        /*c_mod=*/(short)0, c, /*reuse_a=*/false, /*reuse_b=*/false);
}

__global__ __launch_bounds__(256, 2)
void qmha_cos_gemm_kernel(const float* __restrict__ x,
                          const float* __restrict__ theta,
                          const float* __restrict__ W,
                          float* __restrict__ y) {
    // [buf][hi=0/lo=1][...]
    __shared__ __align__(16) _Float16 sA[2][2][BM * LDA];
    __shared__ __align__(16) _Float16 sB[2][2][BN * LDB];

    const int tid   = threadIdx.x;
    const int lane  = tid & 31;
    const int wave  = tid >> 5;        // 0..7
    const int wm    = wave & 3;        // 4 M-groups of 32 rows
    const int wn    = wave >> 2;       // 2 N-groups of 32 cols
    const int lh    = lane & 15;
    const int sel   = (lane >> 4) & 1; // lane half selector

    const int mTile = blockIdx.x;      // 0..63   (feature slice mTile*8..+8)
    const int nTile = blockIdx.y;      // 0..7
    const int batch = blockIdx.z;      // 0..7

    const float INV2PI = 0.15915494309189535f;

    float th[8];
#pragma unroll
    for (int j = 0; j < 8; ++j) th[j] = theta[j];

    // x slice base: x[batch, s, mTile*8 + j] ; A[r][k] = slice[r*64 + k/8][k&7]
    const float* xb = x + (size_t)batch * kS * kE + (size_t)mTile * kDk;

    // --- staging helpers (thread-local raw loads; cos/split done at store) ---
    const int ar = tid >> 1;           // A-row 0..127
    const int ap = tid & 1;            // which pair of seq-rows
    const int wrow = tid >> 2;         // W row 0..63
    const int wc   = tid & 3;          // 8-float chunk within BK

    auto stage_load = [&](int it, v4f xr[4], v4f wr[2]) {
        const float* a0 = xb + (size_t)(ar * 64 + it * 4 + 2 * ap) * kE;
        xr[0] = *(const v4f*)(a0);
        xr[1] = *(const v4f*)(a0 + 4);
        xr[2] = *(const v4f*)(a0 + kE);
        xr[3] = *(const v4f*)(a0 + kE + 4);
        const float* w0 = W + (size_t)(nTile * BN + wrow) * kE + it * BK + wc * 8;
        wr[0] = *(const v4f*)(w0);
        wr[1] = *(const v4f*)(w0 + 4);
    };

    auto stage_store = [&](int buf, const v4f xr[4], const v4f wr[2]) {
        // A: cos(x+theta), split hi/lo f16
#pragma unroll
        for (int q = 0; q < 2; ++q) {
            v8h hi8, lo8;
#pragma unroll
            for (int j = 0; j < 8; ++j) {
                float xv = (j < 4) ? xr[2 * q][j] : xr[2 * q + 1][j - 4];
                float cv = __builtin_amdgcn_cosf((xv + th[j]) * INV2PI);
                _Float16 h = (_Float16)cv;
                hi8[j] = h;
                lo8[j] = (_Float16)(cv - (float)h);
            }
            const int off = ar * LDA + (2 * ap + q) * 8;
            *(v8h*)&sA[buf][0][off] = hi8;
            *(v8h*)&sA[buf][1][off] = lo8;
        }
        // W: split hi/lo f16 (stored as [N][K] — W is already [out,in])
        {
            v8h hi8, lo8;
#pragma unroll
            for (int j = 0; j < 8; ++j) {
                float wv = (j < 4) ? wr[0][j] : wr[1][j - 4];
                _Float16 h = (_Float16)wv;
                hi8[j] = h;
                lo8[j] = (_Float16)(wv - (float)h);
            }
            const int off = wrow * LDB + wc * 8;
            *(v8h*)&sB[buf][0][off] = hi8;
            *(v8h*)&sB[buf][1][off] = lo8;
        }
    };

    // --- prologue: stage K-chunk 0 into buffer 0 ---
    {
        v4f xr[4]; v4f wr[2];
        stage_load(0, xr, wr);
        stage_store(0, xr, wr);
    }
    __syncthreads();

    v8f acc[2][2];
#pragma unroll
    for (int f = 0; f < 2; ++f)
#pragma unroll
        for (int g = 0; g < 2; ++g)
#pragma unroll
            for (int v = 0; v < 8; ++v) acc[f][g][v] = 0.0f;

    // --- main K loop, double-buffered ---
    for (int it = 0; it < KTILES; ++it) {
        const int buf = it & 1;

        v4f nxr[4]; v4f nwr[2];
        const bool more = (it + 1 < KTILES);
        if (more) stage_load(it + 1, nxr, nwr);   // long-latency global loads first

        // Fragment loads (documented 16-bit WMMA VGPR layouts):
        // A (16x32): lanes 0-15 hold K{0-7,16-23}, lanes 16-31 hold K{8-15,24-31}
        // B (32x16): lanes 0-15 hold K{0-7,16-23}, lanes 16-31 hold K{8-15,24-31}
        v16h Ahi[2], Alo[2], Bhi[2], Blo[2];
#pragma unroll
        for (int f = 0; f < 2; ++f) {
            const int row = wm * 32 + f * 16 + lh;
            const int ka  = sel ? 8 : 0;
            const _Float16* ph = &sA[buf][0][row * LDA + ka];
            const _Float16* pl = &sA[buf][1][row * LDA + ka];
            ((v8h*)&Ahi[f])[0] = *(const v8h*)(ph);
            ((v8h*)&Ahi[f])[1] = *(const v8h*)(ph + 16);
            ((v8h*)&Alo[f])[0] = *(const v8h*)(pl);
            ((v8h*)&Alo[f])[1] = *(const v8h*)(pl + 16);
        }
#pragma unroll
        for (int g = 0; g < 2; ++g) {
            const int col = wn * 32 + g * 16 + lh;
            const int kb  = sel ? 8 : 0;
            const _Float16* ph = &sB[buf][0][col * LDB + kb];
            const _Float16* pl = &sB[buf][1][col * LDB + kb];
            ((v8h*)&Bhi[g])[0] = *(const v8h*)(ph);
            ((v8h*)&Bhi[g])[1] = *(const v8h*)(ph + 16);
            ((v8h*)&Blo[g])[0] = *(const v8h*)(pl);
            ((v8h*)&Blo[g])[1] = *(const v8h*)(pl + 16);
        }

        // Split-precision MAC: hi*hi + hi*lo + lo*hi  (~fp32 accuracy)
#pragma unroll
        for (int f = 0; f < 2; ++f)
#pragma unroll
            for (int g = 0; g < 2; ++g) {
                acc[f][g] = wmma_f16(Ahi[f], Bhi[g], acc[f][g]);
                acc[f][g] = wmma_f16(Ahi[f], Blo[g], acc[f][g]);
                acc[f][g] = wmma_f16(Alo[f], Bhi[g], acc[f][g]);
            }

        if (more) stage_store(buf ^ 1, nxr, nwr); // convert+store into other buffer
        __syncthreads();
    }

    // --- epilogue: C/D layout — VGPR v: lanes 0-15 M=v, lanes 16-31 M=v+8 ---
    float* yb = y + ((size_t)batch * kS + (size_t)mTile * BM) * kE + (size_t)nTile * BN;
#pragma unroll
    for (int f = 0; f < 2; ++f)
#pragma unroll
        for (int g = 0; g < 2; ++g) {
            const int col   = wn * 32 + g * 16 + lh;
            const int rbase = wm * 32 + f * 16 + (sel ? 8 : 0);
#pragma unroll
            for (int v = 0; v < 8; ++v)
                yb[(size_t)(rbase + v) * kE + col] = acc[f][g][v];
        }
}

extern "C" void kernel_launch(void* const* d_in, const int* in_sizes, int n_in,
                              void* d_out, int out_size, void* d_ws, size_t ws_size,
                              hipStream_t stream) {
    const float* x     = (const float*)d_in[0];   // (8, 8192, 512) fp32
    const float* theta = (const float*)d_in[1];   // (8,) fp32
    const float* W     = (const float*)d_in[2];   // (512, 512) fp32
    float*       y     = (float*)d_out;           // (8, 8192, 512) fp32

    dim3 grid(kS / BM, kE / BN, kB);              // (64, 8, 8)
    dim3 block(256);                              // 8 waves (wave32)
    hipLaunchKernelGGL(qmha_cos_gemm_kernel, grid, block, 0, stream, x, theta, W, y);
}
